// STULayer_14173392077000
// MI455X (gfx1250) — matrix operand order repaired and checked by reading, not compiled, not run on hardware
//
#include <hip/hip_runtime.h>

#define T_SEQ   2048
#define BATCH   2
#define DMODEL  1024
#define NHEAD   16
#define HDIM    128
#define UVQK_DIM 8192
#define CAT_DIM  3072
#define NROWS   (T_SEQ * BATCH)   // 4096

typedef __attribute__((ext_vector_type(16))) __bf16 v16bf;
typedef __attribute__((ext_vector_type(8)))  __bf16 v8bf;
typedef __attribute__((ext_vector_type(8)))  float  v8f;

union BF16x16 { v16bf v; v8bf h[2]; };

__device__ inline v8f vzero8() {
  v8f v = {0.f,0.f,0.f,0.f,0.f,0.f,0.f,0.f};
  return v;
}

__device__ inline __bf16 f2bf(float f) {
  unsigned u = __float_as_uint(f);
  u += 0x7FFFu + ((u >> 16) & 1u);          // round-to-nearest-even
  unsigned short s = (unsigned short)(u >> 16);
  __bf16 b; __builtin_memcpy(&b, &s, 2); return b;
}
__device__ inline float bf2f(__bf16 b) {
  unsigned short s; __builtin_memcpy(&s, &b, 2);
  return __uint_as_float(((unsigned)s) << 16);
}

// ---------------------------------------------------------------- weight transpose+convert
// in: f32 [K][N] row-major  ->  out: bf16 [N][K] row-major (coalesced both ways)
__global__ __launch_bounds__(256)
void transpose_convert_kernel(const float* __restrict__ in, __bf16* __restrict__ out,
                              int K, int N) {
  __shared__ float tile[32][33];
  const int n0 = blockIdx.x * 32, k0 = blockIdx.y * 32;
  const int tx = threadIdx.x, ty = threadIdx.y;
#pragma unroll
  for (int j = 0; j < 32; j += 8)
    tile[ty + j][tx] = in[(size_t)(k0 + ty + j) * N + n0 + tx];
  __syncthreads();
#pragma unroll
  for (int j = 0; j < 32; j += 8)
    out[(size_t)(n0 + ty + j) * K + k0 + tx] = f2bf(tile[tx][ty + j]);
}

// ---------------------------------------------------------------- V transpose
// uvqk bf16 [T*B][8192], V at col 2048+head*128+d  ->  vt [b*16+head][128][T]
__global__ __launch_bounds__(256)
void transpose_v_kernel(const __bf16* __restrict__ uvqk, __bf16* __restrict__ vt) {
  __shared__ __bf16 tile[32][33];
  const int bh = blockIdx.z;                 // b*16 + head
  const int b = bh >> 4, head = bh & 15;
  const int t0 = blockIdx.x * 32, d0 = blockIdx.y * 32;
  const int tx = threadIdx.x, ty = threadIdx.y;
#pragma unroll
  for (int j = 0; j < 32; j += 8) {
    int t = t0 + ty + j;
    tile[ty + j][tx] =
        uvqk[((size_t)t * BATCH + b) * UVQK_DIM + NHEAD * HDIM + head * HDIM + d0 + tx];
  }
  __syncthreads();
#pragma unroll
  for (int j = 0; j < 32; j += 8) {
    int d = d0 + ty + j;
    vt[((size_t)bh * HDIM + d) * T_SEQ + t0 + tx] = tile[tx][ty + j];
  }
}

// ---------------------------------------------------------------- LN1
__global__ __launch_bounds__(256)
void ln1_kernel(const float* __restrict__ src, const float* __restrict__ w,
                const float* __restrict__ bsh, float* __restrict__ xnorm,
                __bf16* __restrict__ catbuf) {
  __shared__ float red[256];
  const int row = blockIdx.x;
  const int tid = threadIdx.x;
  const float* x = src + (size_t)row * DMODEL;
  float v[4]; float s = 0.f;
#pragma unroll
  for (int i = 0; i < 4; i++) { v[i] = x[tid + i * 256]; s += v[i]; }
  red[tid] = s; __syncthreads();
  for (int off = 128; off > 0; off >>= 1) { if (tid < off) red[tid] += red[tid + off]; __syncthreads(); }
  float mu = red[0] * (1.0f / DMODEL);
  __syncthreads();
  s = 0.f;
#pragma unroll
  for (int i = 0; i < 4; i++) { float d = v[i] - mu; s += d * d; }
  red[tid] = s; __syncthreads();
  for (int off = 128; off > 0; off >>= 1) { if (tid < off) red[tid] += red[tid + off]; __syncthreads(); }
  float rstd = rsqrtf(red[0] * (1.0f / DMODEL) + 1e-5f);
#pragma unroll
  for (int i = 0; i < 4; i++) {
    int c = tid + i * 256;
    float y = (v[i] - mu) * rstd * w[c] + bsh[c];
    xnorm[(size_t)row * DMODEL + c] = y;
    catbuf[(size_t)row * CAT_DIM + 2048 + c] = f2bf(y);
  }
}

// ---------------------------------------------------------------- GEMM
// C[M,N] = A[M,K](bf16) @ Wt[N,K](bf16, pre-transposed) + bias.
// Block tile 128x128, 8 waves, wave = 32x64 (2x4 16x16 WMMA tiles).
template <int OUT_F32>
__global__ __launch_bounds__(256)
void gemm_bf16_kernel(const __bf16* __restrict__ A, int lda,
                      const __bf16* __restrict__ Wt, int N, int K,
                      const float* __restrict__ bias,
                      __bf16* __restrict__ Cbf, float* __restrict__ Cf, int ldc) {
  __shared__ __align__(64) __bf16 As[128][32];   // [m][k]
  __shared__ __align__(64) __bf16 Bt[128][32];   // [n][k] (Wt tile, straight copy)
  const int tid = threadIdx.x;
  const int lane = tid & 31, wid = tid >> 5;
  const int lh = lane >> 4, ln = lane & 15;
  const int m0 = blockIdx.y * 128, n0 = blockIdx.x * 128;
  const int m0w = (wid >> 1) * 32, n0w = (wid & 1) * 64;

  // per-thread staging coords (128x32 tile = 512 v8bf chunks, 2 per thread)
  const int srow0 = tid >> 2, sc0 = (tid & 3) * 8;
  const int srow1 = (tid + 256) >> 2, sc1 = ((tid + 256) & 3) * 8;

  v8f acc[2][4];
#pragma unroll
  for (int i = 0; i < 2; i++)
#pragma unroll
    for (int j = 0; j < 4; j++) acc[i][j] = vzero8();

  for (int k0 = 0; k0 < K; k0 += 32) {
    *(v8bf*)&As[srow0][sc0] = *(const v8bf*)(A + (size_t)(m0 + srow0) * lda + k0 + sc0);
    *(v8bf*)&As[srow1][sc1] = *(const v8bf*)(A + (size_t)(m0 + srow1) * lda + k0 + sc1);
    *(v8bf*)&Bt[srow0][sc0] = *(const v8bf*)(Wt + (size_t)(n0 + srow0) * K + k0 + sc0);
    *(v8bf*)&Bt[srow1][sc1] = *(const v8bf*)(Wt + (size_t)(n0 + srow1) * K + k0 + sc1);
    if (k0 + 32 < K) {                         // prefetch next k-tile
      __builtin_prefetch(A + (size_t)(m0 + srow0) * lda + k0 + 32 + sc0, 0, 1);
      __builtin_prefetch(Wt + (size_t)(n0 + srow0) * K + k0 + 32 + sc0, 0, 1);
    }
    __syncthreads();
    BF16x16 af[2];
#pragma unroll
    for (int mt = 0; mt < 2; ++mt) {           // A fragment: contiguous K half (k = 16*lh + s)
      int r = m0w + mt * 16 + ln;
      int base = lh * 16;
      af[mt].h[0] = *(const v8bf*)&As[r][base];
      af[mt].h[1] = *(const v8bf*)&As[r][base + 8];
    }
#pragma unroll
    for (int nt = 0; nt < 4; ++nt) {
      int c = n0w + nt * 16 + ln;
      v16bf bf = *(const v16bf*)&Bt[c][lh * 16];  // B fragment: contiguous K half
#pragma unroll
      for (int mt = 0; mt < 2; ++mt) {
        acc[mt][nt] = __builtin_amdgcn_wmma_f32_16x16x32_bf16(
            false, af[mt].v, false, bf, (short)0, acc[mt][nt], false, false);
      }
    }
    __syncthreads();
  }
#pragma unroll
  for (int mt = 0; mt < 2; ++mt)
#pragma unroll
    for (int nt = 0; nt < 4; ++nt) {
      int col = n0 + n0w + nt * 16 + ln;
      float bv = bias[col];
#pragma unroll
      for (int r = 0; r < 8; r++) {
        int row = m0 + m0w + mt * 16 + lh * 8 + r;
        float v = acc[mt][nt][r] + bv;
        if (OUT_F32) Cf[(size_t)row * ldc + col] = v;
        else         Cbf[(size_t)row * ldc + col] = f2bf(v);
      }
    }
}

// ---------------------------------------------------------------- attention
// Flash-style causal attention. 4 waves/block; each wave owns 16 query rows.
// Q fragments VGPR-resident; K fragments load directly from global (row-major
// K rows == B^T fragment layout); V fragments load directly from pre-transposed
// global Vt; P redistributed through a tiny wave-private LDS stage.
__global__ __launch_bounds__(128)
void attn_kernel(const __bf16* __restrict__ uvqk, const __bf16* __restrict__ vt,
                 __bf16* __restrict__ catbuf) {
  __shared__ __align__(64) __bf16 Pst[4][16][32];   // [wave][qrow][key]
  const int tid = threadIdx.x;
  const int lane = tid & 31, wid = tid >> 5;
  const int lh = lane >> 4, ln = lane & 15;
  const int b = blockIdx.y >> 4;
  const int head = blockIdx.y & 15;
  const int bh = blockIdx.y;
  const int q0 = blockIdx.x * 64 + wid * 16;
  const float alpha = 0.08838834764831845f;         // 1/sqrt(128)

  const size_t rstride = UVQK_DIM;
  const size_t qcol = 2 * NHEAD * HDIM + (size_t)head * HDIM;
  const size_t kcol = 3 * NHEAD * HDIM + (size_t)head * HDIM;
  const size_t ucol = (size_t)head * HDIM;
  const __bf16* vtp = vt + (size_t)bh * HDIM * T_SEQ;

  BF16x16 qf[4];
  {
    const __bf16* qp = uvqk + ((size_t)(q0 + ln) * BATCH + b) * rstride + qcol;
#pragma unroll
    for (int c = 0; c < 4; c++) {
      int base = c * 32 + lh * 16;
      qf[c].h[0] = *(const v8bf*)(qp + base);
      qf[c].h[1] = *(const v8bf*)(qp + base + 8);
    }
  }

  float mrun[8], lrun[8];
  v8f acc[8];
#pragma unroll
  for (int r = 0; r < 8; r++) { mrun[r] = -__builtin_inff(); lrun[r] = 0.0f; }
#pragma unroll
  for (int d = 0; d < 8; d++) acc[d] = vzero8();

  const int nst = (q0 + 15) / 32 + 1;
  for (int st = 0; st < nst; ++st) {
    const int s0 = st * 32;
    v8f S[2]; S[0] = vzero8(); S[1] = vzero8();
#pragma unroll
    for (int kt = 0; kt < 2; ++kt) {
      const __bf16* kp = uvqk + ((size_t)(s0 + kt * 16 + ln) * BATCH + b) * rstride + kcol;
#pragma unroll
      for (int c = 0; c < 4; c++) {
        v16bf kf = *(const v16bf*)(kp + c * 32 + lh * 16);
        S[kt] = __builtin_amdgcn_wmma_f32_16x16x32_bf16(
            false, qf[c].v, false, kf, (short)0, S[kt], false, false);
      }
    }
#pragma unroll
    for (int r = 0; r < 8; r++) {
      int rowg = q0 + lh * 8 + r;
      float s0v = S[0][r] * alpha, s1v = S[1][r] * alpha;
      if (s0 + ln      > rowg) s0v = -__builtin_inff();
      if (s0 + 16 + ln > rowg) s1v = -__builtin_inff();
      float tmax = fmaxf(s0v, s1v);
#pragma unroll
      for (int off = 8; off >= 1; off >>= 1) tmax = fmaxf(tmax, __shfl_xor(tmax, off, 16));
      float mnew = fmaxf(mrun[r], tmax);
      float scl = __expf(mrun[r] - mnew);
      float p0 = __expf(s0v - mnew);
      float p1 = __expf(s1v - mnew);
      float rs = p0 + p1;
#pragma unroll
      for (int off = 8; off >= 1; off >>= 1) rs += __shfl_xor(rs, off, 16);
      lrun[r] = lrun[r] * scl + rs;
      mrun[r] = mnew;
#pragma unroll
      for (int d = 0; d < 8; ++d) acc[d][r] *= scl;
      Pst[wid][lh * 8 + r][ln]      = f2bf(p0);
      Pst[wid][lh * 8 + r][16 + ln] = f2bf(p1);
    }
    BF16x16 pf;
    pf.h[0] = *(const v8bf*)&Pst[wid][ln][lh * 16];
    pf.h[1] = *(const v8bf*)&Pst[wid][ln][lh * 16 + 8];
#pragma unroll
    for (int d = 0; d < 8; ++d) {
      // B fragment straight from global Vt: row = dim, contiguous keys
      v16bf vf = *(const v16bf*)(vtp + (size_t)(d * 16 + ln) * T_SEQ + s0 + lh * 16);
      acc[d] = __builtin_amdgcn_wmma_f32_16x16x32_bf16(
          false, pf.v, false, vf, (short)0, acc[d], false, false);
    }
  }
  // epilogue: residual = U + attn_out -> cat buffer columns [0,2048)
#pragma unroll
  for (int d = 0; d < 8; ++d) {
    int col = head * HDIM + d * 16 + ln;
#pragma unroll
    for (int r = 0; r < 8; r++) {
      size_t ridx = (size_t)(q0 + lh * 8 + r) * BATCH + b;
      float u = bf2f(uvqk[ridx * rstride + ucol + d * 16 + ln]);
      float o = acc[d][r] / lrun[r] + u;
      catbuf[ridx * CAT_DIM + col] = f2bf(o);
    }
  }
}

// ---------------------------------------------------------------- LN2 + residual
__global__ __launch_bounds__(256)
void ln2_kernel(const float* __restrict__ pre, const float* __restrict__ w,
                const float* __restrict__ bsh, const float* __restrict__ xnorm,
                float* __restrict__ out) {
  __shared__ float red[256];
  const int row = blockIdx.x;
  const int tid = threadIdx.x;
  const float* x = pre + (size_t)row * DMODEL;
  float v[4]; float s = 0.f;
#pragma unroll
  for (int i = 0; i < 4; i++) { v[i] = x[tid + i * 256]; s += v[i]; }
  red[tid] = s; __syncthreads();
  for (int off = 128; off > 0; off >>= 1) { if (tid < off) red[tid] += red[tid + off]; __syncthreads(); }
  float mu = red[0] * (1.0f / DMODEL);
  __syncthreads();
  s = 0.f;
#pragma unroll
  for (int i = 0; i < 4; i++) { float d = v[i] - mu; s += d * d; }
  red[tid] = s; __syncthreads();
  for (int off = 128; off > 0; off >>= 1) { if (tid < off) red[tid] += red[tid + off]; __syncthreads(); }
  float rstd = rsqrtf(red[0] * (1.0f / DMODEL) + 1e-5f);
#pragma unroll
  for (int i = 0; i < 4; i++) {
    int c = tid + i * 256;
    float y = (v[i] - mu) * rstd * w[c] + bsh[c] + xnorm[(size_t)row * DMODEL + c];
    out[(size_t)row * DMODEL + c] = y;
  }
}

// ---------------------------------------------------------------- launch
extern "C" void kernel_launch(void* const* d_in, const int* in_sizes, int n_in,
                              void* d_out, int out_size, void* d_ws, size_t ws_size,
                              hipStream_t stream) {
  const float* src   = (const float*)d_in[0];
  const float* ln1w  = (const float*)d_in[1];
  const float* ln1b  = (const float*)d_in[2];
  const float* wuvqk = (const float*)d_in[3];
  const float* buvqk = (const float*)d_in[4];
  const float* wout  = (const float*)d_in[5];
  const float* bout  = (const float*)d_in[6];
  const float* ln2w  = (const float*)d_in[7];
  const float* ln2b  = (const float*)d_in[8];
  float* out = (float*)d_out;

  char* ws = (char*)d_ws;
  size_t off = 0;
  __bf16* wuvqk_t = (__bf16*)(ws + off); off += (size_t)DMODEL * UVQK_DIM * 2;   // [8192][1024]
  __bf16* wout_t  = (__bf16*)(ws + off); off += (size_t)CAT_DIM * DMODEL * 2;    // [1024][3072]
  __bf16* catbuf  = (__bf16*)(ws + off); off += (size_t)NROWS * CAT_DIM * 2;
  float*  xnorm   = (float*)(ws + off);  off += (size_t)NROWS * DMODEL * 4;
  __bf16* uvqk    = (__bf16*)(ws + off); off += (size_t)NROWS * UVQK_DIM * 2;
  __bf16* vtbuf   = (__bf16*)(ws + off); off += (size_t)BATCH * NHEAD * HDIM * T_SEQ * 2;
  float*  pre     = (float*)(ws + off);  off += (size_t)NROWS * DMODEL * 4;
  (void)ws_size; (void)in_sizes; (void)n_in; (void)out_size;

  // pre-transpose weights to bf16 [N][K]
  transpose_convert_kernel<<<dim3(UVQK_DIM / 32, DMODEL / 32), dim3(32, 8), 0, stream>>>(
      wuvqk, wuvqk_t, DMODEL, UVQK_DIM);
  transpose_convert_kernel<<<dim3(DMODEL / 32, CAT_DIM / 32), dim3(32, 8), 0, stream>>>(
      wout, wout_t, CAT_DIM, DMODEL);

  ln1_kernel<<<NROWS, 256, 0, stream>>>(src, ln1w, ln1b, xnorm, catbuf);

  // uvqk = src_norm @ w_uvqk + b  (A = bf16 src_norm in cat tail, lda=3072)
  gemm_bf16_kernel<0><<<dim3(UVQK_DIM / 128, NROWS / 128), 256, 0, stream>>>(
      catbuf + 2048, CAT_DIM, wuvqk_t, UVQK_DIM, DMODEL, buvqk,
      uvqk, nullptr, UVQK_DIM);

  // V -> Vt[b*16+h][128][T] once (kills per-q-block LDS transposes)
  transpose_v_kernel<<<dim3(T_SEQ / 32, HDIM / 32, BATCH * NHEAD), dim3(32, 8), 0, stream>>>(
      uvqk, vtbuf);

  attn_kernel<<<dim3(T_SEQ / 64, BATCH * NHEAD), 128, 0, stream>>>(uvqk, vtbuf, catbuf);

  // pre = cat @ w_out + b_out
  gemm_bf16_kernel<1><<<dim3(DMODEL / 128, NROWS / 128), 256, 0, stream>>>(
      catbuf, CAT_DIM, wout_t, DMODEL, CAT_DIM, bout,
      nullptr, pre, DMODEL);

  ln2_kernel<<<NROWS, 256, 0, stream>>>(pre, ln2w, ln2b, xnorm, out);
}
